// ALSHConv_85031762526321
// MI455X (gfx1250) — matrix operand order repaired and checked by reading, not compiled, not run on hardware
//
#include <hip/hip_runtime.h>
#include <hip/hip_bf16.h>

#define NH     8
#define BITS   5
#define TS     64
#define M_AUG  27
#define U_SC   0.83f
#define CIN    64
#define COUT   256
#define HIMG   56
#define WIMG   56
#define NB     32
#define KVD    576      // 64*9
#define AD     603      // 576+27
#define HW     3136
#define OUT_ELEMS (NB*COUT*HW)   // 25690112

typedef __bf16 bf16;
typedef __attribute__((ext_vector_type(16))) __bf16 v16bf;
typedef __attribute__((ext_vector_type(8)))  float  v8f;
typedef int v4i_vec __attribute__((vector_size(16)));

// ---- gfx1250 async global->LDS path (ASYNCcnt), with safe fallback ----------
#if __has_builtin(__builtin_amdgcn_global_load_async_to_lds_b128)
#define HAVE_ASYNC_LDS 1
#define ASYNC_B128(gptr, lptr)                                                  \
    __builtin_amdgcn_global_load_async_to_lds_b128(                             \
        (__attribute__((address_space(1))) v4i_vec*)(void*)(gptr),              \
        (__attribute__((address_space(3))) v4i_vec*)(void*)(lptr), 0, 0)
#if __has_builtin(__builtin_amdgcn_s_wait_asynccnt)
#define WAIT_ASYNC() __builtin_amdgcn_s_wait_asynccnt(0)
#else
#define WAIT_ASYNC() asm volatile("s_wait_asynccnt 0x0" ::: "memory")
#endif
#else
#define HAVE_ASYNC_LDS 0
#define WAIT_ASYNC() ((void)0)
#endif

// ---------------- kernel prep: f32 kernels -> bf16 [COUT][KVD] ----------------
__global__ void alsh_prep_b(const float* __restrict__ kern, bf16* __restrict__ Bp) {
    int i = blockIdx.x * 256 + threadIdx.x;
    if (i < COUT * KVD) Bp[i] = (bf16)kern[i];
}

// ---------------- hash the 256 kernel vectors (scale-under-U + ALSH P()) ------
__global__ __launch_bounds__(256) void alsh_hash_kernels(
    const float* __restrict__ kern, const float* __restrict__ a,
    int* __restrict__ bucketk) {
    __shared__ float red[256];
    int k = threadIdx.x;
    const float* kv = kern + k * KVD;
    float norm2 = 0.f;
    for (int d = 0; d < KVD; ++d) { float v = kv[d]; norm2 += v * v; }
    red[k] = sqrtf(norm2);
    __syncthreads();
    for (int off = 128; off > 0; off >>= 1) {
        if (k < off) red[k] = fmaxf(red[k], red[k + off]);
        __syncthreads();
    }
    float scale = U_SC / red[0];
    float aug[M_AUG];
    float p = scale * scale * norm2;          // ||scaled v||^2
    for (int i = 0; i < M_AUG; ++i) { aug[i] = p; p = p * p; }
    int bits[NH * BITS];
    for (int j = 0; j < NH * BITS; ++j) {
        const float* ar = a + j * AD;
        float acc = 0.f;
        for (int d = 0; d < KVD; ++d) acc += scale * kv[d] * ar[d];
        for (int i = 0; i < M_AUG; ++i) acc += aug[i] * ar[KVD + i];
        bits[j] = (acc > 0.f) ? 1 : 0;
    }
    for (int h = 0; h < NH; ++h) {
        int b = 0;
        for (int t = 0; t < BITS; ++t) b += bits[h * BITS + t] << t;
        bucketk[h * COUT + k] = b % TS;
    }
}

// ---------------- zero tallied ------------------------------------------------
__global__ void alsh_zero(int* __restrict__ t) {
    int i = blockIdx.x * 256 + threadIdx.x;
    if (i < NH * TS) t[i] = 0;
}

// ---------------- vote conv + SRP bucket + tally (fused) ----------------------
// grid = (13, NB*NH): h is block-uniform -> weight loads become scalar (SMEM)
__global__ __launch_bounds__(256) void alsh_vote(
    const float* __restrict__ x, const float* __restrict__ a,
    int* __restrict__ tallied) {
    int s = blockIdx.x * 256 + threadIdx.x;
    if (s >= HW) return;
    int h = blockIdx.y & (NH - 1);
    int n = blockIdx.y >> 3;
    int oy = s / WIMG, ox = s - oy * WIMG;
    float acc[BITS] = {0.f, 0.f, 0.f, 0.f, 0.f};
    const float* xb = x + (size_t)n * CIN * HW;
    for (int c = 0; c < CIN; ++c) {
        const float* xc = xb + c * HW;
        for (int ky = 0; ky < 3; ++ky) {
            int iy = oy + ky - 1; if ((unsigned)iy >= HIMG) continue;
            for (int kx = 0; kx < 3; ++kx) {
                int ix = ox + kx - 1; if ((unsigned)ix >= WIMG) continue;
                float xv = xc[iy * WIMG + ix];
                int t = c * 9 + ky * 3 + kx;
#pragma unroll
                for (int b = 0; b < BITS; ++b) acc[b] += xv * a[(h * BITS + b) * AD + t];
            }
        }
    }
    // three constant 0.5 planes (channels 64..66), honoring zero padding
    for (int c = CIN; c < CIN + 3; ++c) {
        for (int ky = 0; ky < 3; ++ky) {
            int iy = oy + ky - 1; if ((unsigned)iy >= HIMG) continue;
            for (int kx = 0; kx < 3; ++kx) {
                int ix = ox + kx - 1; if ((unsigned)ix >= WIMG) continue;
                int t = c * 9 + ky * 3 + kx;
#pragma unroll
                for (int b = 0; b < BITS; ++b) acc[b] += 0.5f * a[(h * BITS + b) * AD + t];
            }
        }
    }
    int bucket = 0;
#pragma unroll
    for (int b = 0; b < BITS; ++b) bucket += (acc[b] > 0.f) << b;
    atomicAdd(&tallied[h * TS + (bucket % TS)], 1);
}

// ---------------- argmax per hash + tallied tail copy -------------------------
__global__ void alsh_argmax(const int* __restrict__ tallied,
                            int* __restrict__ indices,
                            float* __restrict__ out_tail) {
    int h = threadIdx.x;
    if (h < NH) {
        int best = 0, bv = tallied[h * TS];
        for (int i = 1; i < TS; ++i) {
            int v = tallied[h * TS + i];
            if (v > bv) { bv = v; best = i; }       // ties -> lowest index
        }
        indices[h] = best;
        out_tail[NH * TS + h] = (float)best;
    }
    for (int i = threadIdx.x; i < NH * TS; i += blockDim.x)
        out_tail[i] = (float)tallied[i];
}

// ---------------- mask: k active iff bucket_k[h'][k] in {indices} -------------
__global__ void alsh_mask(const int* __restrict__ bucketk,
                          const int* __restrict__ indices,
                          float* __restrict__ maskp,
                          float* __restrict__ out_tail) {
    int k = threadIdx.x;                            // 256 threads
    int m = 0;
    for (int hp = 0; hp < NH; ++hp) {
        int b = bucketk[hp * COUT + k];
        for (int h = 0; h < NH; ++h) if (b == indices[h]) m = 1;
    }
    maskp[k] = (float)m;
    out_tail[NH * TS + NH + k] = (float)m;
}

// ---------------- main conv: implicit GEMM, bf16 WMMA, async double-buffer ----
// block = 256 thr = 8 waves (4M x 2N); tile 64M x 128N; K loop 18 x 32
#define ASTRIDE 40   // bf16 elems per A row (80 B, 16B-aligned)
#define BSTRIDE 56   // bf16 elems per B row (112 B, 16B-aligned, conflict-free)
__global__ __launch_bounds__(256) void alsh_conv_wmma(
    const float* __restrict__ x, const bf16* __restrict__ Bp,
    const float* __restrict__ maskp, float* __restrict__ out) {
    __shared__ bf16 As[2][64 * ASTRIDE];
    __shared__ bf16 Bs[2][128 * BSTRIDE];
    const int tid    = threadIdx.x;
    const int Mblock = blockIdx.x >> 1;
    const int Nblock = blockIdx.x & 1;
    const int Mbase  = Mblock * 64;
    const int Nbase  = Nblock * 128;
    const int lane   = tid & 31;
    const int wave   = tid >> 5;
    const int wm     = wave & 3;       // M sub-tile
    const int wn     = wave >> 2;      // N half (64 cols each)

    // ---- hoisted im2col decode: thread covers 4 (m, kpair) cells -------------
    int mloc[4], kpA[4], oyA[4], oxA[4];
    const float* xbase[4];
#pragma unroll
    for (int i = 0; i < 4; ++i) {
        int e2 = tid + i * 256;                    // 1024 (m, kpair) cells
        int m  = e2 >> 4;
        kpA[i] = (e2 & 15) * 2;
        mloc[i] = m;
        int mg = Mbase + m;
        int n  = mg / HW;
        int s  = mg - n * HW;
        oyA[i] = s / WIMG;
        oxA[i] = s - oyA[i] * WIMG;
        xbase[i] = x + (size_t)n * CIN * HW;
    }
    // ---- hoisted B-staging decode: thread covers 2 b128 transfers ------------
    int nnB[2], partB[2];
#pragma unroll
    for (int i = 0; i < 2; ++i) {
        int q = tid + i * 256;                     // 512 b128 transfers
        nnB[i]   = q >> 2;
        partB[i] = (q & 3) * 8;                    // bf16 element offset
    }

    unsigned apack[4];

    auto loadA = [&](int kk0) {                    // global f32 -> packed bf16 regs
#pragma unroll
        for (int i = 0; i < 4; ++i) {
            int kg = kk0 + kpA[i];
            float v0 = 0.f, v1 = 0.f;
            int c0 = kg / 9, r0 = kg - c0 * 9;
            int ky0 = r0 / 3, kx0 = r0 - ky0 * 3;
            int iy0 = oyA[i] + ky0 - 1, ix0 = oxA[i] + kx0 - 1;
            if ((unsigned)iy0 < HIMG && (unsigned)ix0 < WIMG)
                v0 = xbase[i][(size_t)c0 * HW + iy0 * WIMG + ix0];
            int kg1 = kg + 1;
            int c1 = kg1 / 9, r1 = kg1 - c1 * 9;
            int ky1 = r1 / 3, kx1 = r1 - ky1 * 3;
            int iy1 = oyA[i] + ky1 - 1, ix1 = oxA[i] + kx1 - 1;
            if ((unsigned)iy1 < HIMG && (unsigned)ix1 < WIMG)
                v1 = xbase[i][(size_t)c1 * HW + iy1 * WIMG + ix1];
            unsigned short u0 = __builtin_bit_cast(unsigned short, (bf16)v0);
            unsigned short u1 = __builtin_bit_cast(unsigned short, (bf16)v1);
            apack[i] = (unsigned)u0 | ((unsigned)u1 << 16);
        }
    };
    auto storeA = [&](int buf) {
#pragma unroll
        for (int i = 0; i < 4; ++i)
            ((unsigned*)&As[buf][mloc[i] * ASTRIDE])[kpA[i] >> 1] = apack[i];
    };
    auto issueB = [&](int kk0, int buf) {
#pragma unroll
        for (int i = 0; i < 2; ++i) {
            const bf16* g = Bp + (size_t)(Nbase + nnB[i]) * KVD + kk0 + partB[i];
            bf16* l = &Bs[buf][nnB[i] * BSTRIDE + partB[i]];
#if HAVE_ASYNC_LDS
            ASYNC_B128(g, l);
#else
            *(uint4*)l = *(const uint4*)g;
#endif
        }
    };

    v8f acc[4];
#pragma unroll
    for (int j = 0; j < 4; ++j)
#pragma unroll
        for (int r = 0; r < 8; ++r) acc[j][r] = 0.f;

    loadA(0);
    issueB(0, 0);

    const int mrow  = wm * 16 + (lane & 15);
    const int kbase = (lane >> 4) * 16;            // A: 0 or 16
    const int kb2   = (lane >> 4) * 16;            // B: 0 or 16

    for (int kt = 0; kt < 18; ++kt) {
        const int buf = kt & 1;
        WAIT_ASYNC();                              // B(kt) landed in LDS
        storeA(buf);
        __syncthreads();                           // tiles visible; other buf free
        if (kt + 1 < 18) {
            issueB((kt + 1) * 32, buf ^ 1);        // overlap with WMMAs below
            loadA((kt + 1) * 32);
        }
        // A fragment (contiguous k = 16*(lane/16) + s, as B), two b128 LDS loads
        union { v16bf v; uint4 q[2]; } fa;
        const char* ra = (const char*)&As[buf][mrow * ASTRIDE];
        fa.q[0] = *(const uint4*)(ra + (kbase >> 1) * 4);
        fa.q[1] = *(const uint4*)(ra + ((kbase >> 1) + 4) * 4);
#pragma unroll
        for (int j = 0; j < 4; ++j) {
            const int nrow = wn * 64 + j * 16 + (lane & 15);
            union { v16bf v; uint4 q[2]; } fb;
            const char* rb = (const char*)&Bs[buf][nrow * BSTRIDE];
            fb.q[0] = *(const uint4*)(rb + (kb2 >> 1) * 4);
            fb.q[1] = *(const uint4*)(rb + ((kb2 >> 1) + 4) * 4);
            acc[j] = __builtin_amdgcn_wmma_f32_16x16x32_bf16(
                false, fa.v, false, fb.v, (short)0, acc[j], false, false);
        }
        __syncthreads();
    }

    // store D tiles, fused mask multiply; 8 consecutive pixels per lane
    const int mlocal = wm * 16 + 8 * (lane >> 4);
    const int mg     = Mbase + mlocal;
    const int n_img  = mg / HW;
    const int sbase  = mg - n_img * HW;
#pragma unroll
    for (int j = 0; j < 4; ++j) {
        const int og = Nbase + wn * 64 + j * 16 + (lane & 15);
        const float mk = maskp[og];
        float* op = out + ((size_t)n_img * COUT + og) * HW + sbase;
        float4 lo = make_float4(acc[j][0] * mk, acc[j][1] * mk, acc[j][2] * mk, acc[j][3] * mk);
        float4 hi = make_float4(acc[j][4] * mk, acc[j][5] * mk, acc[j][6] * mk, acc[j][7] * mk);
        *(float4*)(op)     = lo;
        *(float4*)(op + 4) = hi;
    }
}

extern "C" void kernel_launch(void* const* d_in, const int* in_sizes, int n_in,
                              void* d_out, int out_size, void* d_ws, size_t ws_size,
                              hipStream_t stream) {
    (void)in_sizes; (void)n_in; (void)out_size; (void)ws_size;
    const float* x    = (const float*)d_in[0];   // [32,64,56,56]
    const float* kern = (const float*)d_in[1];   // [256,64,3,3]
    const float* a    = (const float*)d_in[2];   // [40,603]
    float* out = (float*)d_out;
    char*  ws  = (char*)d_ws;

    bf16*  Bp      = (bf16*)ws;                  // 294912 B
    int*   bucketk = (int*)(ws + 294912);        // 8*256*4 = 8192 B
    int*   tallied = (int*)(ws + 303104);        // 512*4   = 2048 B
    int*   indices = (int*)(ws + 305152);        // 8*4
    float* maskp   = (float*)(ws + 305216);      // 256*4
    float* out_tail = out + OUT_ELEMS;           // tallied(512) | indices(8) | mask(256)

    alsh_prep_b<<<(COUT * KVD + 255) / 256, 256, 0, stream>>>(kern, Bp);
    alsh_hash_kernels<<<1, 256, 0, stream>>>(kern, a, bucketk);
    alsh_zero<<<2, 256, 0, stream>>>(tallied);
    alsh_vote<<<dim3((HW + 255) / 256, NB * NH), 256, 0, stream>>>(x, a, tallied);
    alsh_argmax<<<1, 64, 0, stream>>>(tallied, indices, out_tail);
    alsh_mask<<<1, 256, 0, stream>>>(bucketk, indices, maskp, out_tail);
    alsh_conv_wmma<<<3136, 256, 0, stream>>>(x, Bp, maskp, out);
}
